// TSUnet_32684701122944
// MI455X (gfx1250) — matrix operand order repaired and checked by reading, not compiled, not run on hardware
//
#include <hip/hip_runtime.h>
#include <hip/hip_bf16.h>
#include <stdint.h>
#include <stddef.h>

// ---------------------------------------------------------------------------
// Types for CDNA5 WMMA (wave32, v_wmma_f32_16x16x32_bf16)
// ---------------------------------------------------------------------------
typedef __attribute__((ext_vector_type(16))) __bf16 v16bf;
typedef __attribute__((ext_vector_type(8)))  float  v8f;
typedef __attribute__((ext_vector_type(4)))  float  f4raw;   // POD 16-byte vector

#define NTOT 8192      // total points (B*NPTS)
#define NPB  2048      // points per batch

__device__ __forceinline__ uint16_t f2b(float f) {
  union { float f; uint32_t u; } x; x.f = f;
  uint32_t u = x.u;
  u += 0x7FFFu + ((u >> 16) & 1u);          // round-to-nearest-even
  return (uint16_t)(u >> 16);
}
__device__ __forceinline__ float b2f(uint16_t h) {
  union { uint32_t u; float f; } x; x.u = ((uint32_t)h) << 16; return x.f;
}

// Flat __shared__ pointer -> wave-relative LDS byte offset (ISA 10.2: LDS
// aperture addresses carry the LDS offset in addr[31:0]).
__device__ __forceinline__ unsigned lds_off(const void* p) {
  return (unsigned)(uintptr_t)p;
}
// CDNA5 async copy: 16B global -> LDS per lane, tracked by ASYNCcnt.
__device__ __forceinline__ void async_ld_b128(unsigned ldsoff, const void* gaddr) {
  asm volatile("global_load_async_to_lds_b128 %0, %1, off"
               :: "v"(ldsoff), "v"(gaddr) : "memory");
}
__device__ __forceinline__ void wait_async0() {
  asm volatile("s_wait_asynccnt 0" ::: "memory");
}

// ---------------------------------------------------------------------------
// Generic bf16 GEMM:  C[N x M] = A[N x K] * W[K x M]  (+bias, +res)
// Workgroup tile 128x64, BK=32, 256 threads = 8 waves, wave tile 32x32 (2x2 WMMA)
// A/B tiles staged via GLOBAL_LOAD_ASYNC_TO_LDS_B128.
// ---------------------------------------------------------------------------
__global__ __launch_bounds__(256) void gemm_bf16_kernel(
    const uint16_t* __restrict__ A, const uint16_t* __restrict__ W,
    int K, int M,
    const float* __restrict__ bias, const float* __restrict__ res,
    float* __restrict__ Cf, uint16_t* __restrict__ Cb)
{
  __shared__ uint16_t As[128][40];   // 40 = 32 + pad (80B rows, 16B aligned)
  __shared__ uint16_t Bs[32][72];    // 72 = 64 + pad (144B rows, 16B aligned)

  const int t    = threadIdx.x;
  const int lane = t & 31, wave = t >> 5;
  const int wr = wave & 3, wc = wave >> 2;       // wave grid 4x2
  const int rowBase = blockIdx.y * 128;
  const int colBase = blockIdx.x * 64;
  const int l16 = lane & 15;
  const int kc  = (lane < 16) ? 0 : 8;           // A-fragment K chunk base

  v8f acc[2][2];
  {
    v8f z = {0.f,0.f,0.f,0.f,0.f,0.f,0.f,0.f};
    acc[0][0]=z; acc[0][1]=z; acc[1][0]=z; acc[1][1]=z;
  }

  const int ar = t >> 1, aseg = (t & 1) * 16;    // A loader: 2 x 16B per thread
  const int bk = t >> 3, bseg = (t & 7) * 8;     // B loader: 1 x 16B per thread
  const unsigned aoff = lds_off(&As[ar][aseg]);
  const unsigned boff = lds_off(&Bs[bk][bseg]);
  const uint16_t* aptr = A + (size_t)(rowBase + ar) * K + aseg;
  const uint16_t* wptr = W + (size_t)bk * M + colBase + bseg;

  for (int k0 = 0; k0 < K; k0 += 32) {
    // speculative prefetch of next tiles (safe past end: dropped on xlat fail)
    __builtin_prefetch(aptr + k0 + 32, 0, 1);
    __builtin_prefetch(wptr + (size_t)(k0 + 32) * M, 0, 1);

    // async stage tiles straight into LDS (ASYNCcnt)
    async_ld_b128(aoff,      aptr + k0);
    async_ld_b128(aoff + 16, aptr + k0 + 8);
    async_ld_b128(boff,      wptr + (size_t)k0 * M);
    wait_async0();
    __syncthreads();

    union Frag { v16bf v; f4raw f[2]; };
    Frag af[2], bf[2];
#pragma unroll
    for (int tr = 0; tr < 2; ++tr) {
      const int row = wr*32 + tr*16 + l16;
      af[tr].f[0] = *(const f4raw*)(&As[row][kc]);
      af[tr].f[1] = *(const f4raw*)(&As[row][kc + 16]);
    }
#pragma unroll
    for (int tc = 0; tc < 2; ++tc) {
      const int col = wc*32 + tc*16;
      for (int s = 0; s < 16; ++s)
        ((uint16_t*)&bf[tc])[s] = Bs[kc + (s & 7) + (s >> 3) * 16][col + l16];
    }
#pragma unroll
    for (int tr = 0; tr < 2; ++tr)
#pragma unroll
      for (int tc = 0; tc < 2; ++tc)
        acc[tr][tc] = __builtin_amdgcn_wmma_f32_16x16x32_bf16(
            false, af[tr].v, false, bf[tc].v, (short)0, acc[tr][tc], false, false);
    __syncthreads();
  }

#pragma unroll
  for (int tr = 0; tr < 2; ++tr) {
#pragma unroll
    for (int tc = 0; tc < 2; ++tc) {
      const int colA = colBase + wc*32 + tc*16 + l16;
      const float bi = bias ? bias[colA] : 0.f;
#pragma unroll
      for (int v = 0; v < 8; ++v) {
        const int rowA = rowBase + wr*32 + tr*16 + v + 8*(lane >> 4);
        const size_t idx = (size_t)rowA * M + colA;
        float val = acc[tr][tc][v] + bi;
        if (res) val += res[idx];
        if (Cf) Cf[idx] = val;
        if (Cb) Cb[idx] = f2b(val);
      }
    }
  }
}

// ---------------------------------------------------------------------------
// Flash attention (B=4, H=4, n=2048, d=32).  qkv bf16 [N x 384] (q|k|v, h*32+d)
// Block = 128 threads (4 waves); wave = 16 query rows; key tiles of 32.
// V tile staged via async-to-LDS; K tile transposed through VGPRs.
// ---------------------------------------------------------------------------
__global__ __launch_bounds__(128) void flash_attn_kernel(
    const uint16_t* __restrict__ qkv, uint16_t* __restrict__ outB)
{
  __shared__ uint16_t KsT[32][40];      // [dd][key]  (transposed)
  __shared__ uint16_t Vs[32][40];       // [key][dd]
  __shared__ uint16_t Ps[4][16][40];    // per-wave P tile 16x32

  const int t = threadIdx.x, wave = t >> 5, lane = t & 31;
  const int bh = blockIdx.y, b = bh >> 2, h = bh & 3;
  const int qbase = blockIdx.x * 64 + wave * 16;
  const int l16 = lane & 15;
  const int kc  = (lane < 16) ? 0 : 8;
  const float sc = 0.1767766952966369f;   // 1/sqrt(32)

  union Frag { v16bf v; f4raw f[2]; };
  Frag aq;
  {
    const size_t g = (size_t)(b*NPB + qbase + l16) * 384 + h*32;
    aq.f[0] = *(const f4raw*)(qkv + g + kc);
    aq.f[1] = *(const f4raw*)(qkv + g + kc + 16);
  }

  float m_[8], l_[8];
  v8f o0 = {0.f,0.f,0.f,0.f,0.f,0.f,0.f,0.f};
  v8f o1 = o0;
#pragma unroll
  for (int v = 0; v < 8; ++v) { m_[v] = -1e30f; l_[v] = 0.f; }

  const int kr = t >> 2;               // key row loader index (0..31)
  const int sseg = (t & 3) * 8;        // dd segment
  const unsigned voff = lds_off(&Vs[kr][sseg]);

  for (int jt = 0; jt < NPB; jt += 32) {
    {  // cooperative K (transposed through VGPRs) + V (async) loads
      const size_t g = (size_t)(b*NPB + jt + kr) * 384 + 128 + h*32 + sseg;
      async_ld_b128(voff, qkv + g + 128);                 // V: mem -> LDS direct
      union { f4raw v; uint16_t u[8]; } kv;
      kv.v = *(const f4raw*)(qkv + g);
#pragma unroll
      for (int j = 0; j < 8; ++j) KsT[sseg + j][kr] = kv.u[j];
      wait_async0();
    }
    __syncthreads();

    Frag bk0, bk1;
    for (int s = 0; s < 16; ++s) {
      ((uint16_t*)&bk0)[s] = KsT[kc + (s & 7) + (s >> 3) * 16][l16];
      ((uint16_t*)&bk1)[s] = KsT[kc + (s & 7) + (s >> 3) * 16][l16 + 16];
    }

    v8f z = {0.f,0.f,0.f,0.f,0.f,0.f,0.f,0.f};
    v8f s0 = __builtin_amdgcn_wmma_f32_16x16x32_bf16(false, aq.v, false, bk0.v, (short)0, z, false, false);
    v8f s1 = __builtin_amdgcn_wmma_f32_16x16x32_bf16(false, aq.v, false, bk1.v, (short)0, z, false, false);

    // online softmax update (row reductions across 16-lane halves)
#pragma unroll
    for (int v = 0; v < 8; ++v) {
      float tm = fmaxf(s0[v], s1[v]);
#pragma unroll
      for (int off = 1; off < 16; off <<= 1) tm = fmaxf(tm, __shfl_xor(tm, off, 32));
      tm *= sc;
      const float mn = fmaxf(m_[v], tm);
      const float alpha = __expf(m_[v] - mn);
      m_[v] = mn;
      const float p0 = __expf(s0[v]*sc - mn);
      const float p1 = __expf(s1[v]*sc - mn);
      s0[v] = p0; s1[v] = p1;
      float rs = p0 + p1;
#pragma unroll
      for (int off = 1; off < 16; off <<= 1) rs += __shfl_xor(rs, off, 32);
      l_[v] = l_[v]*alpha + rs;
      o0[v] *= alpha; o1[v] *= alpha;
    }

    // C-layout -> A-layout transpose of P through LDS (intra-wave)
    const int mrow0 = 8*(lane >> 4);
#pragma unroll
    for (int v = 0; v < 8; ++v) {
      Ps[wave][mrow0 + v][l16]      = f2b(s0[v]);
      Ps[wave][mrow0 + v][l16 + 16] = f2b(s1[v]);
    }
    asm volatile("s_wait_dscnt 0" ::: "memory");   // CDNA5 split DS counter

    Frag ap, bv0, bv1;
    ap.f[0]  = *(const f4raw*)(&Ps[wave][l16][kc]);
    ap.f[1]  = *(const f4raw*)(&Ps[wave][l16][kc + 16]);
    for (int s = 0; s < 16; ++s) {
      ((uint16_t*)&bv0)[s] = Vs[kc + (s & 7) + (s >> 3) * 16][l16];
      ((uint16_t*)&bv1)[s] = Vs[kc + (s & 7) + (s >> 3) * 16][l16 + 16];
    }

    o0 = __builtin_amdgcn_wmma_f32_16x16x32_bf16(false, ap.v, false, bv0.v, (short)0, o0, false, false);
    o1 = __builtin_amdgcn_wmma_f32_16x16x32_bf16(false, ap.v, false, bv1.v, (short)0, o1, false, false);
    __syncthreads();
  }

#pragma unroll
  for (int v = 0; v < 8; ++v) {
    const int m = qbase + v + 8*(lane >> 4);
    const float inv = 1.0f / l_[v];
    const size_t g = (size_t)(b*NPB + m) * 128 + h*32;
    outB[g + l16]      = f2b(o0[v] * inv);
    outB[g + 16 + l16] = f2b(o1[v] * inv);
  }
}

// ---------------------------------------------------------------------------
// Supporting VALU kernels
// ---------------------------------------------------------------------------
__global__ void f32_to_bf16_kernel(const float* __restrict__ s, uint16_t* __restrict__ d, size_t n) {
  size_t i = (size_t)blockIdx.x * 256 + threadIdx.x;
  if (i < n) d[i] = f2b(s[i]);
}

// column batch-norm stats -> affine [a, b2]: y' = a*y + b2
__global__ void colstats_kernel(const float* __restrict__ Y, int N, int M,
                                const float* __restrict__ g, const float* __restrict__ b,
                                float* __restrict__ ab) {
  __shared__ float s1[4][64], s2[4][64];
  const int t = threadIdx.x;
  const int c = blockIdx.x * 64 + (t & 63);
  const int sl = t >> 6;
  float su = 0.f, sq = 0.f;
  for (int r = sl; r < N; r += 4) { float v = Y[(size_t)r*M + c]; su += v; sq += v*v; }
  s1[sl][t & 63] = su; s2[sl][t & 63] = sq;
  __syncthreads();
  if (sl == 0) {
    su = s1[0][t] + s1[1][t] + s1[2][t] + s1[3][t];
    sq = s2[0][t] + s2[1][t] + s2[2][t] + s2[3][t];
    const float mu  = su / N;
    const float var = sq / N - mu*mu;
    const float a   = g[c] * rsqrtf(var + 1e-5f);
    ab[c] = a; ab[M + c] = b[c] - mu*a;
  }
}

// y = silu( a*y + b2  [FiLM: *(1+scale)+shift] ) [+ res]; write f32/bf16
__global__ void epilogue_kernel(const float* __restrict__ Y, const float* __restrict__ ab,
                                const float* __restrict__ te, const int* __restrict__ bidx,
                                const float* __restrict__ res,
                                float* __restrict__ outF, uint16_t* __restrict__ outB,
                                int N, int M) {
  const size_t idx = (size_t)blockIdx.x * 256 + threadIdx.x;
  if (idx >= (size_t)N * M) return;
  const int r = (int)(idx / M), c = (int)(idx % M);
  float v = ab[c] * Y[idx] + ab[M + c];
  if (te) {
    const int bi = bidx[r];
    v = v * (1.f + te[(size_t)bi*2*M + c]) + te[(size_t)bi*2*M + M + c];
  }
  v = v / (1.f + __expf(-v));
  if (res)  v += res[idx];
  if (outF) outF[idx] = v;
  if (outB) outB[idx] = f2b(v);
}

__global__ void concat_bf16_kernel(const float* __restrict__ A, int Ca,
                                   const float* __restrict__ Bs, int Cb,
                                   uint16_t* __restrict__ out, int N) {
  const int Ct = Ca + Cb;
  const size_t idx = (size_t)blockIdx.x * 256 + threadIdx.x;
  if (idx >= (size_t)N * Ct) return;
  const int r = (int)(idx / Ct), c = (int)(idx % Ct);
  const float v = (c < Ca) ? A[(size_t)r*Ca + c] : Bs[(size_t)r*Cb + (c - Ca)];
  out[idx] = f2b(v);
}

// RMSNorm prenorm: xn = x / max(||x||,1e-12) * g * sqrt(C)   (one block per row)
__global__ void rmsnorm_kernel(const float* __restrict__ X, const float* __restrict__ g,
                               uint16_t* __restrict__ outB, int C) {
  __shared__ float red[256];
  const int r = blockIdx.x, t = threadIdx.x;
  float s = 0.f;
  for (int c = t; c < C; c += 256) { float v = X[(size_t)r*C + c]; s += v*v; }
  red[t] = s; __syncthreads();
  for (int off = 128; off > 0; off >>= 1) { if (t < off) red[t] += red[t + off]; __syncthreads(); }
  const float scl = 32.0f / fmaxf(sqrtf(red[0]), 1e-12f);   // sqrt(1024)=32
  for (int c = t; c < C; c += 256) outB[(size_t)r*C + c] = f2b(X[(size_t)r*C + c] * scl * g[c]);
}

__global__ void init_proj_kernel(const float* __restrict__ feats, const float* __restrict__ W,
                                 float* __restrict__ xF, uint16_t* __restrict__ xB) {
  const size_t idx = (size_t)blockIdx.x * 256 + threadIdx.x;
  const int r = (int)(idx >> 7), c = (int)(idx & 127);
  const float v = feats[r*3+0]*W[c] + feats[r*3+1]*W[128+c] + feats[r*3+2]*W[256+c];
  xF[idx] = v; xB[idx] = f2b(v);
}

__global__ void final_proj_kernel(const float* __restrict__ X, const float* __restrict__ W,
                                  float* __restrict__ out) {
  const size_t idx = (size_t)blockIdx.x * 256 + threadIdx.x;
  const int r = (int)(idx / 3), c = (int)(idx % 3);
  float s = 0.f;
  for (int k = 0; k < 128; ++k) s += X[(size_t)r*128 + k] * W[k*3 + c];
  out[idx] = s;
}

__global__ void time_emb_kernel(const float* __restrict__ tm, float* __restrict__ out) {
  const int idx = blockIdx.x * 256 + threadIdx.x;
  if (idx >= 512) return;
  const int bi = idx >> 7, j = idx & 127;
  const float t = tm[bi];
  const float lf = -9.210340371976184f / 63.f;   // -ln(10000)/(half-1)
  out[idx] = (j < 64) ? sinf(t * expf(lf * j)) : cosf(t * expf(lf * (j - 64)));
}

// tiny dense: out[4 x M] = A[4 x K] @ W[K x M] + bias ; act 1 = exact GELU
__global__ void small_gemm_kernel(const float* __restrict__ A, const float* __restrict__ W,
                                  const float* __restrict__ bias, float* __restrict__ out,
                                  int K, int M, int act) {
  const int idx = blockIdx.x * 256 + threadIdx.x;
  if (idx >= 4 * M) return;
  const int r = idx / M, c = idx % M;
  float s = bias ? bias[c] : 0.f;
  for (int k = 0; k < K; ++k) s += A[r*K + k] * W[(size_t)k*M + c];
  if (act == 1) s = 0.5f * s * (1.f + erff(s * 0.7071067811865476f));
  out[idx] = s;
}

__global__ void silu_kernel(const float* __restrict__ in, float* __restrict__ out, int n) {
  const int i = blockIdx.x * 256 + threadIdx.x;
  if (i < n) { float v = in[i]; out[i] = v / (1.f + __expf(-v)); }
}

// ---------------------------------------------------------------------------
// Host orchestration
// ---------------------------------------------------------------------------
extern "C" void kernel_launch(void* const* d_in, const int* in_sizes, int n_in,
                              void* d_out, int out_size, void* d_ws, size_t ws_size,
                              hipStream_t stream)
{
  (void)in_sizes; (void)n_in; (void)out_size; (void)ws_size;
  const float* feats = (const float*)d_in[0];
  const float* timep = (const float*)d_in[1];
  const int*   bidx  = (const int*)d_in[2];
  int cur = 3;
  auto nxt = [&]() { return (const float*)d_in[cur++]; };

  // ---- params (JAX pytree flatten order: sorted dict keys, list order) ----
  const float* attn_Wout = nxt(); const float* attn_Wqkv = nxt();
  const float* attn_bout = nxt(); const float* attn_g    = nxt();

  struct RBp { const float *W1,*W2,*Wres,*b1,*b2,*g1,*g2,*mW,*mb; int din,dout; };
  auto readRB = [&](int din, int dout, bool wres) {
    RBp p; p.din = din; p.dout = dout;
    p.W1 = nxt(); p.W2 = nxt(); p.Wres = wres ? nxt() : nullptr;
    p.b1 = nxt(); p.b2 = nxt(); p.g1 = nxt(); p.g2 = nxt(); p.mW = nxt(); p.mb = nxt();
    return p;
  };

  const float* downW[4]; RBp drb1[4], drb2[4];
  const int ddin[4]  = {128,128,256,512}, ddout[4] = {128,256,512,1024};
  for (int i = 0; i < 4; ++i) { downW[i]=nxt(); drb1[i]=readRB(ddin[i],ddin[i],false); drb2[i]=readRB(ddin[i],ddin[i],false); }
  const float* finalW = nxt();
  RBp frb  = readRB(256, 128, true);
  const float* initW  = nxt();
  RBp mrb1 = readRB(1024, 1024, false);
  RBp mrb2 = readRB(1024, 1024, false);
  const float* tW1 = nxt(); const float* tW2 = nxt();
  const float* tb1 = nxt(); const float* tb2 = nxt();
  RBp urb1[4], urb2[4]; const float* upW[4];
  const int udin[4] = {1536,768,384,256}, udout[4] = {1024,512,256,128}, uoutc[4] = {512,256,128,128};
  for (int i = 0; i < 4; ++i) { urb1[i]=readRB(udin[i],udout[i],true); urb2[i]=readRB(udin[i],udout[i],true); upW[i]=nxt(); }

  // ---- workspace bump allocator ----
  char* wsp = (char*)d_ws;
  auto alloc  = [&](size_t bytes) { void* p = wsp; wsp += (bytes + 255) & ~(size_t)255; return p; };
  auto allocF = [&](size_t n) { return (float*)alloc(n * sizeof(float)); };
  auto allocB = [&](size_t n) { return (uint16_t*)alloc(n * 2); };
  auto cvt = [&](const float* src, size_t n) {
    uint16_t* d = allocB(n);
    f32_to_bf16_kernel<<<dim3((unsigned)((n + 255) / 256)), dim3(256), 0, stream>>>(src, d, n);
    return d;
  };
  struct RBd { uint16_t *W1, *W2, *Wres; };
  auto mkRB = [&](const RBp& p) {
    RBd d; d.W1 = cvt(p.W1, (size_t)p.din * p.dout); d.W2 = cvt(p.W2, (size_t)p.dout * p.dout);
    d.Wres = p.Wres ? cvt(p.Wres, (size_t)p.din * p.dout) : nullptr; return d;
  };

  RBd drb1d[4], drb2d[4], urb1d[4], urb2d[4], frbd, mrb1d, mrb2d;
  uint16_t *downWd[4], *upWd[4];
  for (int i = 0; i < 4; ++i) { downWd[i] = cvt(downW[i], (size_t)ddin[i]*ddout[i]); drb1d[i] = mkRB(drb1[i]); drb2d[i] = mkRB(drb2[i]); }
  frbd = mkRB(frb); mrb1d = mkRB(mrb1); mrb2d = mkRB(mrb2);
  for (int i = 0; i < 4; ++i) { urb1d[i] = mkRB(urb1[i]); urb2d[i] = mkRB(urb2[i]); upWd[i] = cvt(upW[i], (size_t)udout[i]*uoutc[i]); }
  uint16_t* WqkvB = cvt(attn_Wqkv, (size_t)1024*384);
  uint16_t* WoutB = cvt(attn_Wout, (size_t)128*1024);

  float* t0    = allocF(4*128);
  float* t1    = allocF(4*512);
  float* tfin  = allocF(4*512);
  float* tact  = allocF(4*512);
  float* teBuf = allocF(4*2048);
  float* ab    = allocF(2*1024);

  float* x0F = allocF((size_t)NTOT*128);
  const int hsC[8] = {128,128,128,128,256,256,512,512};
  float* hsF[8]; for (int i = 0; i < 8; ++i) hsF[i] = allocF((size_t)NTOT*hsC[i]);
  float* xF0  = allocF((size_t)NTOT*1024);
  float* xF1  = allocF((size_t)NTOT*1024);
  float* tmpF = allocF((size_t)NTOT*1024);
  float* Ytmp = allocF((size_t)NTOT*1024);
  float* Rtmp = allocF((size_t)NTOT*1024);
  uint16_t* bufB0 = allocB((size_t)NTOT*1024);
  uint16_t* bufB1 = allocB((size_t)NTOT*1024);
  uint16_t* Htmp  = allocB((size_t)NTOT*1024);
  uint16_t* catB  = allocB((size_t)NTOT*1536);
  uint16_t* qkvB  = allocB((size_t)NTOT*384);
  uint16_t* attnB = allocB((size_t)NTOT*128);

  auto gemm = [&](const uint16_t* A, const uint16_t* W, int K, int M,
                  const float* bias, const float* res, float* Cf, uint16_t* Cb) {
    gemm_bf16_kernel<<<dim3(M/64, NTOT/128), dim3(256), 0, stream>>>(A, W, K, M, bias, res, Cf, Cb);
  };
  auto colstats = [&](const float* Y, int M, const float* g, const float* b) {
    colstats_kernel<<<dim3(M/64), dim3(256), 0, stream>>>(Y, NTOT, M, g, b, ab);
  };
  auto epi = [&](const float* Y, int M, const float* te, const float* res, float* oF, uint16_t* oB) {
    epilogue_kernel<<<dim3((unsigned)(((size_t)NTOT*M)/256)), dim3(256), 0, stream>>>(Y, ab, te, bidx, res, oF, oB, NTOT, M);
  };
  auto resnet = [&](const RBp& p, const RBd& d, const uint16_t* inB, const float* inF,
                    float* outF, uint16_t* outB) {
    small_gemm_kernel<<<dim3((4*2*p.dout)/256), dim3(256), 0, stream>>>(tact, p.mW, p.mb, teBuf, 512, 2*p.dout, 0);
    gemm(inB, d.W1, p.din, p.dout, nullptr, nullptr, Ytmp, nullptr);
    colstats(Ytmp, p.dout, p.g1, p.b1);
    epi(Ytmp, p.dout, teBuf, nullptr, nullptr, Htmp);
    gemm(Htmp, d.W2, p.dout, p.dout, nullptr, nullptr, Ytmp, nullptr);
    colstats(Ytmp, p.dout, p.g2, p.b2);
    const float* res = inF;
    if (d.Wres) { gemm(inB, d.Wres, p.din, p.dout, nullptr, nullptr, Rtmp, nullptr); res = Rtmp; }
    epi(Ytmp, p.dout, nullptr, res, outF, outB);
  };
  auto concat = [&](const float* A, int Ca, const float* Bs, int Cb, uint16_t* o) {
    const size_t tot = (size_t)NTOT * (Ca + Cb);
    concat_bf16_kernel<<<dim3((unsigned)(tot/256)), dim3(256), 0, stream>>>(A, Ca, Bs, Cb, o, NTOT);
  };

  // ---- time embedding path ----
  time_emb_kernel<<<dim3(2), dim3(256), 0, stream>>>(timep, t0);
  small_gemm_kernel<<<dim3(8), dim3(256), 0, stream>>>(t0, tW1, tb1, t1, 128, 512, 1);
  small_gemm_kernel<<<dim3(8), dim3(256), 0, stream>>>(t1, tW2, tb2, tfin, 512, 512, 0);
  silu_kernel<<<dim3(8), dim3(256), 0, stream>>>(tfin, tact, 2048);

  // ---- init projection ----
  init_proj_kernel<<<dim3((NTOT*128)/256), dim3(256), 0, stream>>>(feats, initW, x0F, bufB0);

  const uint16_t* curB = bufB0; const float* curF = x0F;
  int k = 0;
  // ---- down path ----
  for (int i = 0; i < 4; ++i) {
    uint16_t* ob1 = (curB == bufB0) ? bufB1 : bufB0;
    resnet(drb1[i], drb1d[i], curB, curF, hsF[k], ob1); curB = ob1; curF = hsF[k]; ++k;
    uint16_t* ob2 = (curB == bufB0) ? bufB1 : bufB0;
    resnet(drb2[i], drb2d[i], curB, curF, hsF[k], ob2); curB = ob2; curF = hsF[k]; ++k;
    uint16_t* ob3 = (curB == bufB0) ? bufB1 : bufB0;
    gemm(curB, downWd[i], ddin[i], ddout[i], nullptr, nullptr, xF0, ob3);
    curF = xF0; curB = ob3;
  }
  // ---- mid rb1 ----
  { uint16_t* ob = (curB == bufB0) ? bufB1 : bufB0;
    resnet(mrb1, mrb1d, curB, curF, xF1, ob); curB = ob; curF = xF1; }
  // ---- attention ----
  rmsnorm_kernel<<<dim3(NTOT), dim3(256), 0, stream>>>(curF, attn_g, catB, 1024);
  gemm(catB, WqkvB, 1024, 384, nullptr, nullptr, nullptr, qkvB);
  flash_attn_kernel<<<dim3(NPB/64, 16), dim3(128), 0, stream>>>(qkvB, attnB);
  { uint16_t* ob = (curB == bufB0) ? bufB1 : bufB0;
    gemm(attnB, WoutB, 128, 1024, attn_bout, curF, xF0, ob);
    curB = ob; curF = xF0; }
  // ---- mid rb2 ----
  { uint16_t* ob = (curB == bufB0) ? bufB1 : bufB0;
    resnet(mrb2, mrb2d, curB, curF, xF1, ob); curB = ob; curF = xF1; }
  // ---- up path ----
  for (int i = 0; i < 4; ++i) {
    --k; concat(curF, udout[i], hsF[k], hsC[k], catB);
    resnet(urb1[i], urb1d[i], catB, nullptr, tmpF, nullptr);
    --k; concat(tmpF, udout[i], hsF[k], hsC[k], catB);
    uint16_t* ob = (curB == bufB0) ? bufB1 : bufB0;
    resnet(urb2[i], urb2d[i], catB, nullptr, tmpF, ob);
    float* nxtF = (curF == xF0) ? xF1 : xF0;
    gemm(ob, upWd[i], udout[i], uoutc[i], nullptr, nullptr, nxtF, nullptr);
    curF = nxtF; curB = ob;
  }
  // ---- final ----
  concat(curF, 128, x0F, 128, catB);
  resnet(frb, frbd, catB, nullptr, tmpF, nullptr);
  final_proj_kernel<<<dim3((NTOT*3)/256), dim3(256), 0, stream>>>(tmpF, finalW, (float*)d_out);
}
